// EdgeTransition_60541859004908
// MI455X (gfx1250) — matrix operand order repaired and checked by reading, not compiled, not run on hardware
//
#include <hip/hip_runtime.h>
#include <hip/hip_bf16.h>

typedef __bf16 bf16;
typedef __attribute__((ext_vector_type(16))) __bf16 v16bf;
typedef __attribute__((ext_vector_type(8)))  __bf16 v8bf;
typedef __attribute__((ext_vector_type(8)))  float  v8f;
typedef __attribute__((ext_vector_type(4)))  float  v4f;

#define B_      4
#define N_      1024
#define K_      64
#define NODE_D  256
#define EDGE_IN 128
#define EDGE_OUT 128
#define BIAS_D  128
#define HID     384
#define ROWS    (B_ * N_ * K_)          // 262144
#define NNODES  (B_ * N_)               // 4096
#define MT      64                      // rows per workgroup == one (b,i) node group
#define XSE     136                     // edge slab stride (128 + 8 pad bf16)
#define XST     392                     // t/u slab stride (384 + 8 pad bf16)
#define EPS_    1e-5f

// ---------------- WMMA fragment loaders ----------------
// A fragment (16x32 bf16), ISA 7.12.2 16-bit A layout:
// lanes 0-15: V0..V3 = K 0..7,  V4..V7 = K 16..23  (mrow = lane)
// lanes 16-31: V0..V3 = K 8..15, V4..V7 = K 24..31 (mrow = lane-16)
__device__ inline v16bf load_a_lds(const bf16* slab, int stride, int row16base,
                                   int kbase, int lane) {
    int half = lane >> 4, mrow = lane & 15;
    const bf16* p = slab + (row16base + mrow) * stride + kbase + half * 8;
    v8bf lo = *(const v8bf*)p;
    v8bf hi = *(const v8bf*)(p + 16);
    v16bf a;
#pragma unroll
    for (int i = 0; i < 8; ++i) { a[i] = lo[i]; a[i + 8] = hi[i]; }
    return a;
}

// B fragment (32x16 bf16) from transposed weights Wt[out][in]:
// lane n = lane&15 is the output column; lanes 0-15 hold K 0..7,16..23, lanes 16-31 K 8..15,24..31.
__device__ inline v16bf load_b_glb(const bf16* Wt, int Kdim, int ncol_base,
                                   int kbase, int lane) {
    int n  = lane & 15;
    int ks = kbase + ((lane >> 4) << 3);
    const bf16* p = Wt + (size_t)(ncol_base + n) * Kdim + ks;
    v8bf lo = *(const v8bf*)p;
    v8bf hi = *(const v8bf*)(p + 16);
    v16bf b;
#pragma unroll
    for (int i = 0; i < 8; ++i) { b[i] = lo[i]; b[i + 8] = hi[i]; }
    return b;
}

// A fragment (16x32 bf16) straight from a row-major bf16 global matrix
__device__ inline v16bf load_a_glb_bf16(const bf16* A, int stride, int row16base,
                                        int kbase, int lane) {
    int half = lane >> 4, mrow = lane & 15;
    const bf16* p = A + (size_t)(row16base + mrow) * stride + kbase + half * 8;
    v8bf lo = *(const v8bf*)p;
    v8bf hi = *(const v8bf*)(p + 16);
    v16bf a;
#pragma unroll
    for (int i = 0; i < 8; ++i) { a[i] = lo[i]; a[i + 8] = hi[i]; }
    return a;
}

// ---------------- weight prep: transpose f32 -> bf16 ----------------
__global__ void et_convert_transpose(const float* __restrict__ in, bf16* __restrict__ out,
                                     int IN, int OUT) {
    int idx = blockIdx.x * blockDim.x + threadIdx.x;
    if (idx >= IN * OUT) return;
    int o = idx / IN, i = idx % IN;
    out[idx] = (bf16)in[(size_t)i * OUT + o];
}

// ---------------- h = node_emb @ W0 + b0  (bf16 out) ----------------
__global__ __launch_bounds__(32) void et_node_embed(const float* __restrict__ node_emb,
                                                    const bf16* __restrict__ W0t,
                                                    const float* __restrict__ b0,
                                                    bf16* __restrict__ hb) {
    int mt   = blockIdx.x;               // 16-row tile
    int lane = threadIdx.x & 31;
    int half = lane >> 4, mrow = lane & 15;
    const float* arow = node_emb + (size_t)(mt * 16 + mrow) * NODE_D;

    for (int nt = 0; nt < BIAS_D / 16; ++nt) {
        v8f acc = {};
        for (int kt = 0; kt < NODE_D / 32; ++kt) {
            int kb = kt * 32;
            const float* pa = arow + kb + half * 8;
            v4f f0 = *(const v4f*)pa;
            v4f f1 = *(const v4f*)(pa + 4);
            v4f f2 = *(const v4f*)(pa + 16);
            v4f f3 = *(const v4f*)(pa + 20);
            v16bf a;
#pragma unroll
            for (int j = 0; j < 4; ++j) {
                a[j] = (bf16)f0[j]; a[4 + j] = (bf16)f1[j];
                a[8 + j] = (bf16)f2[j]; a[12 + j] = (bf16)f3[j];
            }
            v16bf b = load_b_glb(W0t, NODE_D, nt * 16, kb, lane);
            acc = __builtin_amdgcn_wmma_f32_16x16x32_bf16(false, a, false, b,
                                                          (short)0, acc, false, false);
        }
        int col = nt * 16 + (lane & 15);
        float bias = b0[col];
#pragma unroll
        for (int r = 0; r < 8; ++r)
            hb[(size_t)(mt * 16 + r + half * 8) * BIAS_D + col] = (bf16)(acc[r] + bias);
    }
}

// ---------------- per-node precompute: out[4096][nout] = hb @ Wt[:, koff:koff+128] ----------------
__global__ __launch_bounds__(32) void et_node_gemm(const bf16* __restrict__ hb,
                                                   const bf16* __restrict__ Wt,
                                                   int wt_kdim, int koff, int nout,
                                                   float* __restrict__ out) {
    int mt   = blockIdx.x;               // 16 node rows
    int lane = threadIdx.x & 31;
    int half = lane >> 4;

    for (int nt = 0; nt < nout / 16; ++nt) {
        v8f acc = {};
        for (int kt = 0; kt < BIAS_D / 32; ++kt) {
            int kb = kt * 32;
            v16bf a = load_a_glb_bf16(hb, BIAS_D, mt * 16, kb, lane);
            v16bf b = load_b_glb(Wt, wt_kdim, nt * 16, koff + kb, lane);
            acc = __builtin_amdgcn_wmma_f32_16x16x32_bf16(false, a, false, b,
                                                          (short)0, acc, false, false);
        }
        int col = nt * 16 + (lane & 15);
#pragma unroll
        for (int r = 0; r < 8; ++r)
            out[(size_t)(mt * 16 + r + half * 8) * nout + col] = acc[r];
    }
}

// ---------------- fused trunk ----------------
// rows of one block share node i (K_=64 == MT); left/right GEMM parts folded into
// precomputed hpL/hpR (layer1) and hfL/hfR (final), gathered as bias vectors.
__global__ __launch_bounds__(64) void et_trunk(const float* __restrict__ edge_emb,
                                               const int*  __restrict__ E_idx,
                                               const bf16* __restrict__ W1t,
                                               const bf16* __restrict__ W2t,
                                               const bf16* __restrict__ Wft,
                                               const float* __restrict__ b1,
                                               const float* __restrict__ b2,
                                               const float* __restrict__ bfv,
                                               const float* __restrict__ hpL,
                                               const float* __restrict__ hpR,
                                               const float* __restrict__ hfL,
                                               const float* __restrict__ hfR,
                                               const float* __restrict__ lng,
                                               const float* __restrict__ lnb,
                                               float* __restrict__ out) {
    extern __shared__ char smem[];
    bf16* xs = (bf16*)smem;                 // 64 x XSE  (edge features, bf16)
    bf16* ts = xs + MT * XSE;               // 64 x XST  (layer1 out; later f32 y stash)
    bf16* us = ts + MT * XST;               // 64 x XST  (layer2 out)
    int*  jidx = (int*)(us + MT * XST);     // 64 gather indices

    const int rowbase = blockIdx.x * MT;
    const int node = rowbase >> 6;          // flat b*N+i (MT == K_)
    const int bN   = (node / N_) * N_;      // b*N
    const int tid  = threadIdx.x;
    const int lane = tid & 31;
    const int wv   = tid >> 5;              // wave 0..1 owns local rows [wv*32, wv*32+32)
    const int r16  = wv * 32;
    const int half = lane >> 4;
    const int n15  = lane & 15;

    // ---- build edge slab (1 thread per row) + gather indices ----
    {
        const float* e = edge_emb + (size_t)(rowbase + tid) * EDGE_IN;
        bf16* xrow = xs + tid * XSE;
        for (int c = 0; c < EDGE_IN; c += 8) {
            v4f f0 = *(const v4f*)(e + c);
            v4f f1 = *(const v4f*)(e + c + 4);
            v8bf o;
#pragma unroll
            for (int j = 0; j < 4; ++j) { o[j] = (bf16)f0[j]; o[4 + j] = (bf16)f1[j]; }
            *(v8bf*)(xrow + c) = o;
        }
        jidx[tid] = E_idx[rowbase + tid];
    }
    __syncthreads();

    // ---- GEMM1': t = relu(edge @ W1[0:128] + hpL[i] + hpR[j] + b1), M-block 2 ----
    for (int nt = 0; nt < HID / 16; ++nt) {
        v8f acc0 = {}, acc1 = {};
        for (int kt = 0; kt < EDGE_IN / 32; ++kt) {
            int kb = kt * 32;
            v16bf a0 = load_a_lds(xs, XSE, r16, kb, lane);
            v16bf a1 = load_a_lds(xs, XSE, r16 + 16, kb, lane);
            v16bf b  = load_b_glb(W1t, HID, nt * 16, kb, lane);
            acc0 = __builtin_amdgcn_wmma_f32_16x16x32_bf16(false, a0, false, b, (short)0, acc0, false, false);
            acc1 = __builtin_amdgcn_wmma_f32_16x16x32_bf16(false, a1, false, b, (short)0, acc1, false, false);
        }
        int col = nt * 16 + n15;
        float base = b1[col] + hpL[(size_t)node * HID + col];
#pragma unroll
        for (int r = 0; r < 8; ++r) {
            int lr = r16 + r + half * 8;
            float v = acc0[r] + base + hpR[(size_t)(bN + jidx[lr]) * HID + col];
            ts[lr * XST + col] = (bf16)(v > 0.f ? v : 0.f);
        }
#pragma unroll
        for (int r = 0; r < 8; ++r) {
            int lr = r16 + 16 + r + half * 8;
            float v = acc1[r] + base + hpR[(size_t)(bN + jidx[lr]) * HID + col];
            ts[lr * XST + col] = (bf16)(v > 0.f ? v : 0.f);
        }
    }

    // ---- GEMM2: u = relu(t @ W2 + b2), K=384, M-block 2 ----
    for (int nt = 0; nt < HID / 16; ++nt) {
        v8f acc0 = {}, acc1 = {};
        for (int kt = 0; kt < HID / 32; ++kt) {
            int kb = kt * 32;
            v16bf a0 = load_a_lds(ts, XST, r16, kb, lane);
            v16bf a1 = load_a_lds(ts, XST, r16 + 16, kb, lane);
            v16bf b  = load_b_glb(W2t, HID, nt * 16, kb, lane);
            acc0 = __builtin_amdgcn_wmma_f32_16x16x32_bf16(false, a0, false, b, (short)0, acc0, false, false);
            acc1 = __builtin_amdgcn_wmma_f32_16x16x32_bf16(false, a1, false, b, (short)0, acc1, false, false);
        }
        int col = nt * 16 + n15;
        float bias = b2[col];
#pragma unroll
        for (int r = 0; r < 8; ++r) {
            float v = acc0[r] + bias;
            us[(r16 + r + half * 8) * XST + col] = (bf16)(v > 0.f ? v : 0.f);
        }
#pragma unroll
        for (int r = 0; r < 8; ++r) {
            float v = acc1[r] + bias;
            us[(r16 + 16 + r + half * 8) * XST + col] = (bf16)(v > 0.f ? v : 0.f);
        }
    }

    // ---- GEMM3': y = u @ Wf + edge @ Wf[0:128] + hfL[i] + hfR[j] + bf ----
    // y (f32) stashed into this wave's dead region of the t slab.
    float* yws = (float*)((char*)ts + wv * (32 * XST * (int)sizeof(bf16))); // 25088 B >= 16384 B
    for (int nt = 0; nt < EDGE_OUT / 16; ++nt) {
        v8f acc0 = {}, acc1 = {};
        for (int kt = 0; kt < HID / 32; ++kt) {          // u part, K=384
            int kb = kt * 32;
            v16bf a0 = load_a_lds(us, XST, r16, kb, lane);
            v16bf a1 = load_a_lds(us, XST, r16 + 16, kb, lane);
            v16bf b  = load_b_glb(Wft, HID, nt * 16, kb, lane);
            acc0 = __builtin_amdgcn_wmma_f32_16x16x32_bf16(false, a0, false, b, (short)0, acc0, false, false);
            acc1 = __builtin_amdgcn_wmma_f32_16x16x32_bf16(false, a1, false, b, (short)0, acc1, false, false);
        }
        for (int kt = 0; kt < EDGE_IN / 32; ++kt) {      // edge residual part, K=128
            int kb = kt * 32;
            v16bf a0 = load_a_lds(xs, XSE, r16, kb, lane);
            v16bf a1 = load_a_lds(xs, XSE, r16 + 16, kb, lane);
            v16bf b  = load_b_glb(Wft, HID, nt * 16, kb, lane);
            acc0 = __builtin_amdgcn_wmma_f32_16x16x32_bf16(false, a0, false, b, (short)0, acc0, false, false);
            acc1 = __builtin_amdgcn_wmma_f32_16x16x32_bf16(false, a1, false, b, (short)0, acc1, false, false);
        }
        int col = nt * 16 + n15;
        float base = bfv[col] + hfL[(size_t)node * EDGE_OUT + col];
#pragma unroll
        for (int r = 0; r < 8; ++r) {
            int lrm = r + half * 8;
            yws[lrm * EDGE_OUT + col] =
                acc0[r] + base + hfR[(size_t)(bN + jidx[r16 + lrm]) * EDGE_OUT + col];
        }
#pragma unroll
        for (int r = 0; r < 8; ++r) {
            int lrm = 16 + r + half * 8;
            yws[lrm * EDGE_OUT + col] =
                acc1[r] + base + hfR[(size_t)(bN + jidx[r16 + lrm]) * EDGE_OUT + col];
        }
    }

    // ---- LayerNorm over 128 cols (wave-cooperative, wave32 shfl) ----
    const int growb = rowbase + r16;
    for (int m = 0; m < 32; ++m) {
        float v0 = yws[m * EDGE_OUT + lane];
        float v1 = yws[m * EDGE_OUT + lane + 32];
        float v2 = yws[m * EDGE_OUT + lane + 64];
        float v3 = yws[m * EDGE_OUT + lane + 96];
        float s  = v0 + v1 + v2 + v3;
        float s2 = v0 * v0 + v1 * v1 + v2 * v2 + v3 * v3;
#pragma unroll
        for (int off = 16; off > 0; off >>= 1) {
            s  += __shfl_xor(s,  off, 32);
            s2 += __shfl_xor(s2, off, 32);
        }
        float mean = s * (1.f / EDGE_OUT);
        float var  = s2 * (1.f / EDGE_OUT) - mean * mean;
        float rstd = rsqrtf(var + EPS_);
        size_t ob = (size_t)(growb + m) * EDGE_OUT;
        out[ob + lane]      = (v0 - mean) * rstd * lng[lane]      + lnb[lane];
        out[ob + lane + 32] = (v1 - mean) * rstd * lng[lane + 32] + lnb[lane + 32];
        out[ob + lane + 64] = (v2 - mean) * rstd * lng[lane + 64] + lnb[lane + 64];
        out[ob + lane + 96] = (v3 - mean) * rstd * lng[lane + 96] + lnb[lane + 96];
    }
}

// ---------------- host launcher ----------------
extern "C" void kernel_launch(void* const* d_in, const int* in_sizes, int n_in,
                              void* d_out, int out_size, void* d_ws, size_t ws_size,
                              hipStream_t stream) {
    const float* node_emb = (const float*)d_in[0];
    const float* edge_emb = (const float*)d_in[1];
    const int*   E_idx    = (const int*)d_in[2];
    const float* W0  = (const float*)d_in[3];
    const float* b0  = (const float*)d_in[4];
    const float* W1  = (const float*)d_in[5];
    const float* b1  = (const float*)d_in[6];
    const float* W2  = (const float*)d_in[7];
    const float* b2  = (const float*)d_in[8];
    const float* Wf  = (const float*)d_in[9];
    const float* bfv = (const float*)d_in[10];
    const float* lng = (const float*)d_in[11];
    const float* lnb = (const float*)d_in[12];
    float* out = (float*)d_out;

    char* ws = (char*)d_ws;
    bf16*  W0t = (bf16*)(ws);                    //   65536 B : [128][256]
    bf16*  W1t = (bf16*)(ws + 65536);            //  294912 B : [384][384]
    bf16*  W2t = (bf16*)(ws + 360448);           //  294912 B : [384][384]
    bf16*  Wft = (bf16*)(ws + 655360);           //   98304 B : [128][384]
    bf16*  hb  = (bf16*)(ws + 753664);           // 1048576 B : [4096][128] bf16
    float* hpL = (float*)(ws + 1802240);         // 6291456 B : [4096][384]
    float* hpR = (float*)(ws + 8093696);         // 6291456 B : [4096][384]
    float* hfL = (float*)(ws + 14385152);        // 2097152 B : [4096][128]
    float* hfR = (float*)(ws + 16482304);        // 2097152 B : [4096][128]  (end ~17.7 MB)

    et_convert_transpose<<<(NODE_D * BIAS_D + 255) / 256, 256, 0, stream>>>(W0, W0t, NODE_D, BIAS_D);
    et_convert_transpose<<<(HID * HID + 255) / 256, 256, 0, stream>>>(W1, W1t, HID, HID);
    et_convert_transpose<<<(HID * HID + 255) / 256, 256, 0, stream>>>(W2, W2t, HID, HID);
    et_convert_transpose<<<(HID * EDGE_OUT + 255) / 256, 256, 0, stream>>>(Wf, Wft, HID, EDGE_OUT);

    et_node_embed<<<NNODES / 16, 32, 0, stream>>>(node_emb, W0t, b0, hb);

    // per-node folded contributions of the "left"/"right" concat features
    et_node_gemm<<<NNODES / 16, 32, 0, stream>>>(hb, W1t, HID, 128, HID, hpL);
    et_node_gemm<<<NNODES / 16, 32, 0, stream>>>(hb, W1t, HID, 256, HID, hpR);
    et_node_gemm<<<NNODES / 16, 32, 0, stream>>>(hb, Wft, HID, 128, EDGE_OUT, hfL);
    et_node_gemm<<<NNODES / 16, 32, 0, stream>>>(hb, Wft, HID, 256, EDGE_OUT, hfR);

    size_t shmem = (size_t)MT * XSE * sizeof(bf16)
                 + 2 * (size_t)MT * XST * sizeof(bf16)
                 + MT * sizeof(int);              // 118016 B
    (void)hipFuncSetAttribute((const void*)et_trunk,
                              hipFuncAttributeMaxDynamicSharedMemorySize, (int)shmem);
    et_trunk<<<ROWS / MT, 64, shmem, stream>>>(edge_emb, E_idx, W1t, W2t, Wft,
                                               b1, b2, bfv, hpL, hpR, hfL, hfR,
                                               lng, lnb, out);
}
